// FullAutoLSTM_69793218560435
// MI455X (gfx1250) — matrix operand order repaired and checked by reading, not compiled, not run on hardware
//
#include <hip/hip_runtime.h>

// ---------------------------------------------------------------------------
// FullAutoLSTM for MI455X (gfx1250, wave32).
//
// Persistent kernel: 64 WGs x 16 batch rows, each runs all 1024 time steps
// (batch rows are independent through the recurrence). Per step:
//   z[16x1024] = [x|h][16x288] @ [W_ih|W_hh][288x1024] + b
// via v_wmma_f32_16x16x32_bf16 (72 WMMAs/wave/step). Weights are packed bf16
// in a per-lane WMMA B layout (contiguous k, as A) and streamed from L2 (0.59MB resident
// in 192MB L2) with a 2-deep software pipeline. The weight base pointer is
// laundered through an asm barrier each step (defeats LICM -> no scratch
// spills) and cast to address_space(1) so the stream uses global_load_b128
// (LOADcnt only) instead of flat_load_b128 (which also ties up DScnt/LDS).
// ---------------------------------------------------------------------------

typedef __attribute__((ext_vector_type(16))) __bf16        v16bf;
typedef __attribute__((ext_vector_type(8)))  float         v8f;
typedef __attribute__((ext_vector_type(4)))  unsigned int  v4u;

union BF16Tile { v4u q[2]; v16bf v; };

typedef __attribute__((address_space(1))) const v4u* gb4p;

__device__ __forceinline__ unsigned short f2bf(float f) {
    unsigned int u = __float_as_uint(f);
    return (unsigned short)((u + 0x7FFFu + ((u >> 16) & 1u)) >> 16);
}
__device__ __forceinline__ float sigmoid_fast(float x) {
    return 1.0f / (1.0f + __expf(-x));
}
__device__ __forceinline__ float tanh_fast(float x) {
    return 1.0f - 2.0f / (__expf(2.0f * x) + 1.0f);
}

// ---------------------------------------------------------------------------
// Prologue: pack W_ih (32x1024) and W_hh (256x1024) fp32 -> bf16 tiles in the
// per-lane B-operand layout used here for v_wmma_f32_16x16x32_bf16 (contiguous k, as A):
//   tile (kt, nt): kt=0 -> W_ih rows 0..31; kt=1..8 -> W_hh rows (kt-1)*32 ..
//   lane L: n = nt*16 + (L&15), K half = L>>4 (K = half*16 .. half*16+15)
//   lane's 16 bf16 stored contiguously (32B) -> two global_load_b128 in loop.
// ---------------------------------------------------------------------------
__global__ void pack_weights(const float* __restrict__ W_ih,
                             const float* __restrict__ W_hh,
                             unsigned short* __restrict__ wpack) {
    int g = blockIdx.x * blockDim.x + threadIdx.x;   // 9*64 tiles * 32 lanes
    if (g >= 9 * 64 * 32) return;
    int lane = g & 31;
    int tile = g >> 5;
    int kt = tile >> 6;          // 0..8
    int nt = tile & 63;          // 0..63
    int n  = nt * 16 + (lane & 15);
    int khalf = lane >> 4;

    v4u d0, d1;
#pragma unroll
    for (int i = 0; i < 8; ++i) {
        int k0 = khalf * 16 + 2 * i;
        float w0, w1;
        if (kt == 0) {
            w0 = W_ih[(k0 + 0) * 1024 + n];
            w1 = W_ih[(k0 + 1) * 1024 + n];
        } else {
            int kb = (kt - 1) * 32;
            w0 = W_hh[(kb + k0 + 0) * 1024 + n];
            w1 = W_hh[(kb + k0 + 1) * 1024 + n];
        }
        unsigned int uu = (unsigned int)f2bf(w0) | ((unsigned int)f2bf(w1) << 16);
        if (i < 4) d0[i] = uu; else d1[i - 4] = uu;
    }
    v4u* d4 = reinterpret_cast<v4u*>(wpack + (size_t)(tile * 32 + lane) * 16);
    d4[0] = d0;
    d4[1] = d1;
}

// ---------------------------------------------------------------------------
// Persistent sequential LSTM kernel. grid=64 (batch tiles of 16), block=256.
// ---------------------------------------------------------------------------
__global__ __launch_bounds__(256, 1)
void lstm_seq(const float* __restrict__ c0, const float* __restrict__ h0,
              const float* __restrict__ xw,   // warmup_input [1024][512][32]
              const float* __restrict__ xa,   // auto_input   [1024][512][24]
              const float* __restrict__ bias, // [1024]
              const float* __restrict__ Wout, // [256][8]
              const float* __restrict__ bout, // [8]
              const unsigned short* __restrict__ wpack,
              float* __restrict__ out)        // [1024][1024][8]
{
    __shared__ float          zbuf[16 * 1024];   // activated gates  (64KB)
    __shared__ float          hbuf[16 * 256];    // h fp32           (16KB)
    __shared__ unsigned short hbf [16 * 256];    // h bf16 row-major ( 8KB)
    __shared__ unsigned short xtile[16 * 32];    // x bf16 row-major ( 1KB)
    __shared__ float          woutT[8 * 260];    // W_out^T, padded  ( 8KB)
    __shared__ float          boutS[8];

    const int tid  = threadIdx.x;
    const int lane = tid & 31;
    const int wv   = tid >> 5;          // wave 0..7
    const int wg   = blockIdx.x;        // batch tile 0..63
    const int nl   = lane & 15;
    const int hi   = lane >> 4;

    // ---- one-time init ----
    // W_out^T: woutT[o*260 + k] = Wout[k*8 + o]; stride 260 (=4 mod 64 banks)
    for (int i = tid; i < 8 * 256; i += 256) {
        int o = i >> 8, k = i & 255;
        woutT[o * 260 + k] = Wout[k * 8 + o];
    }
    if (tid < 8) boutS[tid] = bout[tid];

    const int em = tid >> 4;            // elementwise row 0..15
    const int ej = tid & 15;
    float creg[16];
#pragma unroll
    for (int i = 0; i < 16; ++i) {
        int j  = ej + i * 16;
        int bg = wg * 16 + em;
        creg[i] = c0[(size_t)bg * 256 + j];
        float hv = h0[(size_t)bg * 256 + j];
        hbuf[em * 256 + j] = hv;
        hbf [em * 256 + j] = f2bf(hv);
    }

    float bvals[8];
#pragma unroll
    for (int p = 0; p < 8; ++p) bvals[p] = bias[wv * 128 + p * 16 + nl];

    { // x tile for t=0 (warmup, full 32 features)
        int v = tid * 2, m = v >> 5, cc = v & 31;
        const float* src = xw + ((size_t)(wg * 16 + m) * 512 + 0) * 32 + cc;
        xtile[m * 32 + cc]     = f2bf(src[0]);
        xtile[m * 32 + cc + 1] = f2bf(src[1]);
    }
    __syncthreads();

    const int gate = wv >> 1;           // 0:i 1:f 2:g 3:o (wave-uniform)
    unsigned long long wbase = (unsigned long long)wpack;

    for (int t = 0; t < 1024; ++t) {
        // Launder the weight base pointer: defeats LICM so the weight stream
        // stays inside the loop; AS(1) cast gives global_load_b128.
        asm volatile("" : "+s"(wbase));
        gb4p wp4 = (gb4p)wbase;

        // ---- phase 1: z = x@Wih + h@Whh + b via WMMA, 2-deep B pipeline ----
        v8f acc[8];
#pragma unroll
        for (int p = 0; p < 8; ++p) {
            v8f a;
#pragma unroll
            for (int e = 0; e < 8; ++e) a[e] = bvals[p];
            acc[p] = a;
        }

        BF16Tile Bt[2][8];
#pragma unroll
        for (int p = 0; p < 8; ++p) {   // preload kt=0 B tiles
            size_t idx = ((size_t)(wv * 8 + p) * 64 + lane * 2);
            Bt[0][p].q[0] = wp4[idx + 0];
            Bt[0][p].q[1] = wp4[idx + 1];
        }
#pragma unroll
        for (int kt = 0; kt < 9; ++kt) {
            BF16Tile A;
            if (kt == 0) {
                A.q[0] = *(const v4u*)(xtile + nl * 32 + hi * 16);
                A.q[1] = *(const v4u*)(xtile + nl * 32 + hi * 16 + 8);
            } else {
                int k0 = (kt - 1) * 32;
                A.q[0] = *(const v4u*)(hbf + nl * 256 + k0 + hi * 16);
                A.q[1] = *(const v4u*)(hbf + nl * 256 + k0 + hi * 16 + 8);
            }
            if (kt < 8) {               // prefetch kt+1 B tiles
#pragma unroll
                for (int p = 0; p < 8; ++p) {
                    size_t idx = ((size_t)((kt + 1) * 64 + wv * 8 + p) * 64 + lane * 2);
                    Bt[(kt + 1) & 1][p].q[0] = wp4[idx + 0];
                    Bt[(kt + 1) & 1][p].q[1] = wp4[idx + 1];
                }
            }
#pragma unroll
            for (int p = 0; p < 8; ++p) {
                acc[p] = __builtin_amdgcn_wmma_f32_16x16x32_bf16(
                    false, A.v, false, Bt[kt & 1][p].v, (short)0, acc[p],
                    false, false);
            }
        }

        // ---- phase 2: activations -> zbuf (gate is wave-uniform) ----
#pragma unroll
        for (int p = 0; p < 8; ++p) {
            int n = (wv * 8 + p) * 16 + nl;
#pragma unroll
            for (int e = 0; e < 8; ++e) {
                float z = acc[p][e];
                float a = (gate == 2) ? tanh_fast(z) : sigmoid_fast(z);
                zbuf[(e + 8 * hi) * 1024 + n] = a;
            }
        }
        __syncthreads();

        // ---- phase 3: c,h update (c in VGPRs) ----
#pragma unroll
        for (int i = 0; i < 16; ++i) {
            int j  = ej + i * 16;
            float iv = zbuf[em * 1024 +        j];
            float fv = zbuf[em * 1024 + 256 + j];
            float gv = zbuf[em * 1024 + 512 + j];
            float ov = zbuf[em * 1024 + 768 + j];
            float cv = fv * creg[i] + iv * gv;
            creg[i] = cv;
            float hv = ov * tanh_fast(cv);
            hbuf[em * 256 + j] = hv;
            hbf [em * 256 + j] = f2bf(hv);
        }
        __syncthreads();

        // ---- phase 4: y = h@Wout + bout, store output, build next x ----
        int tn = t + 1;
        if (tn < 1024) {
            if (tn < 512) {                       // warmup: full 32 columns
                int v = tid * 2, m = v >> 5, cc = v & 31;
                const float* src = xw + ((size_t)(wg * 16 + m) * 512 + tn) * 32 + cc;
                xtile[m * 32 + cc]     = f2bf(src[0]);
                xtile[m * 32 + cc + 1] = f2bf(src[1]);
            } else if (tid < 192) {               // auto: columns 0..23
                int v = tid * 2, m = v / 24, cc = v % 24;
                const float* src = xa + ((size_t)(wg * 16 + m) * 512 + (tn - 512)) * 24 + cc;
                xtile[m * 32 + cc]     = f2bf(src[0]);
                xtile[m * 32 + cc + 1] = f2bf(src[1]);
            }
        }
        if (tid < 128) {                          // y head: (m, o) pairs
            int m = tid >> 3, o = tid & 7;
            float yv = boutS[o];
            const float* hrow = hbuf + m * 256;
            const float* wrow = woutT + o * 260;
#pragma unroll 2
            for (int k = 0; k < 256; k += 4) {
                float4 h4 = *(const float4*)(hrow + k);
                float4 w4 = *(const float4*)(wrow + k);
                yv += h4.x * w4.x + h4.y * w4.y + h4.z * w4.z + h4.w * w4.w;
            }
            out[((size_t)(wg * 16 + m) * 1024 + t) * 8 + o] = yv;
            if (tn >= 512 && tn < 1024)           // feedback columns 24..31
                xtile[m * 32 + 24 + o] = f2bf(yv);
        }
        __syncthreads();
    }
}

extern "C" void kernel_launch(void* const* d_in, const int* in_sizes, int n_in,
                              void* d_out, int out_size, void* d_ws, size_t ws_size,
                              hipStream_t stream) {
    const float* c0   = (const float*)d_in[0];
    const float* h0   = (const float*)d_in[1];
    const float* xw   = (const float*)d_in[2];
    const float* xa   = (const float*)d_in[3];
    const float* Wih  = (const float*)d_in[4];
    const float* Whh  = (const float*)d_in[5];
    const float* b    = (const float*)d_in[6];
    const float* Wout = (const float*)d_in[7];
    const float* bout = (const float*)d_in[8];
    unsigned short* wpack = (unsigned short*)d_ws;   // 9*64*32*16 bf16 = 576KB

    pack_weights<<<72, 256, 0, stream>>>(Wih, Whh, wpack);
    lstm_seq<<<64, 256, 0, stream>>>(c0, h0, xw, xa, b, Wout, bout, wpack,
                                     (float*)d_out);
    (void)in_sizes; (void)n_in; (void)out_size; (void)ws_size;
}
